// GraphAttention_49185965474594
// MI455X (gfx1250) — matrix operand order repaired and checked by reading, not compiled, not run on hardware
//
#include <hip/hip_runtime.h>
#include <hip/hip_bf16.h>

typedef __attribute__((ext_vector_type(16))) __bf16 v16bf;
typedef __attribute__((ext_vector_type(8)))  __bf16 v8bf;
typedef __attribute__((ext_vector_type(8)))  float  v8f;
typedef __attribute__((ext_vector_type(4)))  float  v4f;
typedef __attribute__((ext_vector_type(4)))  unsigned int uint32x4;
typedef __attribute__((ext_vector_type(8)))  int    int32x8;
typedef __attribute__((ext_vector_type(4)))  int    int32x4;

constexpr int kBATCH = 256;
constexpr int kCH    = 128;   // rows of h (M)
constexpr int kT     = 512;   // reduction dim of GEMM1 (K1)
constexpr int kHID   = 256;   // cols of h (N)
constexpr int kChunkRows = 32;                 // W rows per staged chunk
constexpr int kChunkElems = kChunkRows * kT;   // 16384 bf16 = 32 KB

__device__ __forceinline__ unsigned short f2bf_rne(float f) {
    unsigned int u = __float_as_uint(f);
    u += 0x7FFFu + ((u >> 16) & 1u);
    return (unsigned short)(u >> 16);
}

// Convert W_w (fp32 -> bf16) into workspace; tiny, runs once, stays in L2.
__global__ void convert_w_kernel(const float* __restrict__ W,
                                 unsigned short* __restrict__ Wb) {
    int i = blockIdx.x * blockDim.x + threadIdx.x;   // 512*256 == 131072 exact
    Wb[i] = f2bf_rne(W[i]);
}

// One workgroup per batch. 256 threads = 8 wave32s; wave w owns M-tile w.
__launch_bounds__(256, 1)
__global__ void gat_fused_kernel(const float* __restrict__ x,
                                 const unsigned short* __restrict__ Wb_,
                                 const float* __restrict__ W_b,
                                 const float* __restrict__ a_w,
                                 const float* __restrict__ a_b,
                                 float* __restrict__ out) {
    __shared__ __bf16 h_t[kHID * kCH];            // 64 KB: h TRANSPOSED [n][m]
    // 64 KB overlay: w_stage (2x32KB, GEMM1 only) then alpha_b (32KB, after)
    __shared__ __align__(16) unsigned char smem_ov[2 * kChunkElems * 2];
    __shared__ float  fvec[kCH], gvec[kCH], colsum[kCH];

    __bf16* w_stage = (__bf16*)smem_ov;           // [2][kChunkElems]
    __bf16* alpha_b = (__bf16*)smem_ov;           // [kCH*kCH], after GEMM1

    const int b       = blockIdx.x;
    const int tid     = threadIdx.x;
    const int wave    = tid >> 5;
    const int lane    = tid & 31;
    const int lane16  = lane & 15;
    const int halfsel = lane >> 4;        // 0 for lanes 0-15, 1 for 16-31

    const __bf16* wmat = (const __bf16*)Wb_;
    const float*  xrow = x + (long long)b * kCH * kT;

    const int m_base = wave * 16;
    const int arow   = m_base + lane16;        // A-matrix row held by this lane
    const int m0     = m_base + halfsel * 8;   // first of this lane's 8 D rows

    // ---------- W chunk staging: TDM double buffer (32 rows / 32 KB) ----------
    auto stage_w = [&](int c, int buf) {
#if __has_builtin(__builtin_amdgcn_tensor_load_to_lds)
        if (tid < 32) {   // one wave issues the DMA (EXEC ignored by TDM)
            unsigned long long ga =
                (unsigned long long)(wmat + (long long)c * kChunkElems);
            unsigned lds_off =
                (unsigned)(unsigned long long)&w_stage[buf * kChunkElems];
            uint32x4 g0;
            g0[0] = 1u;                                   // count = 1
            g0[1] = lds_off;                              // lds_addr (bytes)
            g0[2] = (unsigned)ga;                         // global_addr[31:0]
            g0[3] = (unsigned)((ga >> 32) & 0x01FFFFFFull)// global_addr[56:32]
                    | 0x80000000u;                        // type = 2 ("image")
            // group1: data_size=1(2B); tensor_dim0=16384; tensor_dim1=1;
            //         tile_dim0=16384; tile_dim1=1; strides=16384
            int32x8 g1;
            g1[0] = 0x00010000;   // wg_mask=0, data_size=1 (2 bytes)
            g1[1] = 0x40000000;   // tensor_dim0[15:0]=16384 @ bits 63:48
            g1[2] = 0x00010000;   // tensor_dim0 hi=0, tensor_dim1 lo=1
            g1[3] = 0x40000000;   // tensor_dim1 hi=0, tile_dim0=16384
            g1[4] = 0x00000001;   // tile_dim1=1, tile_dim2=0
            g1[5] = 0x00004000;   // tensor_dim0_stride = 16384
            g1[6] = 0x40000000;   // stride0 hi=0, tensor_dim1_stride lo=16384
            g1[7] = 0x00000000;
            int32x4 z4 = {0, 0, 0, 0};
            int32x8 z8 = {0, 0, 0, 0, 0, 0, 0, 0};
            __builtin_amdgcn_tensor_load_to_lds(g0, g1, z4, z4, z8, 0);
        }
#else
        // cooperative fallback: 256 threads x 128 B
        const __bf16* src = wmat + (long long)c * kChunkElems;
        __bf16* dst = &w_stage[buf * kChunkElems];
        const int base = tid * 64;
        #pragma unroll
        for (int e = 0; e < 8; ++e)
            *(v8bf*)&dst[base + e * 8] = *(const v8bf*)(src + base + e * 8);
#endif
    };
    auto wait_w = [&]() {
#if __has_builtin(__builtin_amdgcn_tensor_load_to_lds)
        if (tid < 32) __builtin_amdgcn_s_wait_tensorcnt(0);
#endif
        __syncthreads();
    };

    // ---- Load this wave's x rows once: fp32 -> bf16 in-register ----
    // A layout: lanes 0-15 hold K {kb..kb+15},
    //           lanes 16-31 hold K {kb+16..kb+31}
    v16bf afr[16];
    #pragma unroll
    for (int ks = 0; ks < 16; ++ks) {
        const float* ap = xrow + (long long)arow * kT + ks * 32 + halfsel * 16;
        __builtin_prefetch(ap + 64, 0, 1);
        v4f l0 = *(const v4f*)(ap);
        v4f l1 = *(const v4f*)(ap + 4);
        v4f h0 = *(const v4f*)(ap + 8);
        v4f h1 = *(const v4f*)(ap + 12);
        v16bf a;
        #pragma unroll
        for (int e = 0; e < 4; ++e) {
            a[e]      = (__bf16)l0[e];
            a[e + 4]  = (__bf16)l1[e];
            a[e + 8]  = (__bf16)h0[e];
            a[e + 12] = (__bf16)h1[e];
        }
        afr[ks] = a;
    }

    // ================= GEMM1: h = x @ W^T + bias =================
    // 8 chunks of 32 W rows; two n-tiles per chunk -> two independent
    // accumulator chains sharing the A fragments (ILP hides LDS latency).
    stage_w(0, 0);
    for (int c = 0; c < 8; ++c) {
        const int buf = c & 1;
        wait_w();                               // chunk c landed in w_stage
        if (c + 1 < 8) stage_w(c + 1, buf ^ 1); // overlap next DMA
        const __bf16* wb    = &w_stage[buf * kChunkElems];
        const __bf16* wrow0 = wb + lane16 * kT;
        const __bf16* wrow1 = wb + (16 + lane16) * kT;
        v8f acc0 = {}, acc1 = {};
        #pragma unroll
        for (int ks = 0; ks < 16; ++ks) {
            v16bf b0 = *(const v16bf*)(wrow0 + ks * 32 + halfsel * 16);
            v16bf b1 = *(const v16bf*)(wrow1 + ks * 32 + halfsel * 16);
            acc0 = __builtin_amdgcn_wmma_f32_16x16x32_bf16(
                    false, afr[ks], false, b0, (short)0, acc0, false, false);
            acc1 = __builtin_amdgcn_wmma_f32_16x16x32_bf16(
                    false, afr[ks], false, b1, (short)0, acc1, false, false);
        }
        const int bn0 = c * 32 + lane16;        // first n-tile column
        const int bn1 = bn0 + 16;               // second n-tile column
        const float bias0 = W_b[bn0];
        const float bias1 = W_b[bn1];
        v8bf hv0, hv1;
        #pragma unroll
        for (int r = 0; r < 8; ++r) {
            hv0[r] = (__bf16)(acc0[r] + bias0);
            hv1[r] = (__bf16)(acc1[r] + bias1);
        }
        // transposed stores: column bn, rows m0..m0+7 contiguous -> b128 each
        *(v8bf*)(&h_t[bn0 * kCH + m0]) = hv0;
        *(v8bf*)(&h_t[bn1 * kCH + m0]) = hv1;
    }
    __syncthreads();   // also ends w_stage lifetime (overlay -> alpha_b)

    // ================= f = h@a_src, g = h@a_dst =================
    if (tid < kCH) {
        float s = 0.f;
        for (int n = 0; n < kHID; ++n)
            s += (float)h_t[n * kCH + tid] * a_w[n];
        fvec[tid] = s;
    } else {
        const int i = tid - kCH;
        float s = 0.f;
        for (int n = 0; n < kHID; ++n)
            s += (float)h_t[n * kCH + i] * a_w[kHID + n];
        gvec[i] = s;
    }
    __syncthreads();

    // ============ e = leaky_relu(f_i + g_j + b); exp (bf16) ============
    const float ab = a_b[0];
    for (int idx = tid; idx < kCH * kCH; idx += 256) {
        const int i = idx >> 7, j = idx & 127;
        float e = fvec[i] + gvec[j] + ab;
        e = (e > 0.f) ? e : 0.2f * e;
        alpha_b[idx] = (__bf16)__expf(e);
    }
    __syncthreads();
    // softmax over axis 1 (sum over i for each column j), fp32 accumulate
    if (tid < kCH) {
        float s = 0.f;
        for (int i = 0; i < kCH; ++i) s += (float)alpha_b[i * kCH + tid];
        colsum[tid] = s;
    }
    __syncthreads();
    for (int idx = tid; idx < kCH * kCH; idx += 256) {
        alpha_b[idx] = (__bf16)((float)alpha_b[idx] * __frcp_rn(colsum[idx & 127]));
    }
    __syncthreads();

    // ================= GEMM2: out = alpha @ h =================
    v16bf afr2[4];
    #pragma unroll
    for (int ks = 0; ks < 4; ++ks) {
        const __bf16* ap = &alpha_b[arow * kCH + ks * 32 + halfsel * 16];
        v8bf alo = *(const v8bf*)(ap);
        v8bf ahi = *(const v8bf*)(ap + 8);
        afr2[ks] = __builtin_shufflevector(alo, ahi,
            0,1,2,3,4,5,6,7,8,9,10,11,12,13,14,15);
    }
    float* orow = out + (long long)b * kCH * kHID;
    // two n-tiles per iteration: two independent WMMA chains here as well
    for (int np = 0; np < 8; ++np) {
        const int N0 = np * 32 + lane16;
        const int N1 = N0 + 16;
        const __bf16* hc0 = &h_t[N0 * kCH];
        const __bf16* hc1 = &h_t[N1 * kCH];
        v8f acc0 = {}, acc1 = {};
        #pragma unroll
        for (int ks = 0; ks < 4; ++ks) {
            v16bf b0 = *(const v16bf*)(hc0 + ks * 32 + halfsel * 16);
            v16bf b1 = *(const v16bf*)(hc1 + ks * 32 + halfsel * 16);
            acc0 = __builtin_amdgcn_wmma_f32_16x16x32_bf16(
                    false, afr2[ks], false, b0, (short)0, acc0, false, false);
            acc1 = __builtin_amdgcn_wmma_f32_16x16x32_bf16(
                    false, afr2[ks], false, b1, (short)0, acc1, false, false);
        }
        #pragma unroll
        for (int r = 0; r < 8; ++r) {
            orow[(m0 + r) * kHID + N0] = acc0[r];
            orow[(m0 + r) * kHID + N1] = acc1[r];
        }
    }
}

extern "C" void kernel_launch(void* const* d_in, const int* in_sizes, int n_in,
                              void* d_out, int out_size, void* d_ws, size_t ws_size,
                              hipStream_t stream) {
    const float* x   = (const float*)d_in[0];   // (256,128,512)
    const float* W_w = (const float*)d_in[1];   // (256,512)
    const float* W_b = (const float*)d_in[2];   // (256,)
    const float* a_w = (const float*)d_in[3];   // (512,)
    const float* a_b = (const float*)d_in[4];   // (1,)

    unsigned short* Wb = (unsigned short*)d_ws;          // 256 KB

    convert_w_kernel<<<512, 256, 0, stream>>>(W_w, Wb);
    gat_fused_kernel<<<kBATCH, 256, 0, stream>>>(x, Wb, W_b, a_w, a_b,
                                                 (float*)d_out);
}
